// Net_661424963757
// MI455X (gfx1250) — matrix operand order repaired and checked by reading, not compiled, not run on hardware
//
#include <hip/hip_runtime.h>

// LIF SNN fused kernels for MI455X (gfx1250, wave32, WMMA + TDM/async-LDS).
// B=4096, NI=1024, NH=2048, NO=256, 25 steps.

#define BETA    0.95f
#define THRESH  1.0f
#define T_STEPS 25
#define Bsz     4096
#define NI      1024
#define NH      2048
#define NO      256

typedef __attribute__((ext_vector_type(16))) __bf16 v16bf;
typedef __attribute__((ext_vector_type(8)))  float  v8f;
typedef __attribute__((ext_vector_type(4)))  unsigned int u32x4;
typedef __attribute__((ext_vector_type(4)))  int i32x4;
typedef __attribute__((ext_vector_type(8)))  int i32x8;

#ifndef __has_builtin
#define __has_builtin(x) 0
#endif

// CDNA5 data movers, in order of preference, each compile-guarded:
//   1) Tensor Data Mover (one wave DMAs the whole 128KB tile, TENSORcnt wait)
//   2) async global->LDS B128 (ASYNCcnt wait)
//   3) plain coalesced loads (always available)
#if defined(__AMDGCN__) && __has_builtin(__builtin_amdgcn_tensor_load_to_lds) && \
    __has_builtin(__builtin_amdgcn_s_wait_tensorcnt)
#define USE_TDM 1
#endif
#if defined(__AMDGCN__) && __has_builtin(__builtin_amdgcn_global_load_async_to_lds_b128) && \
    __has_builtin(__builtin_amdgcn_s_wait_asynccnt)
#define USE_ASYNC 1
#endif

// ---------------------------------------------------------------------------
// Kernel 0: split W2 (f32) into bf16 hi/lo pair (hi+lo ~ f32 to 2^-16 rel).
// ---------------------------------------------------------------------------
__global__ __launch_bounds__(256) void split_w2_kernel(
    const float* __restrict__ W2, __bf16* __restrict__ w2h, __bf16* __restrict__ w2l) {
  int i = blockIdx.x * 256 + threadIdx.x;
  if (i < NO * NH) {
    float v = W2[i];
    __bf16 h = (__bf16)v;
    w2h[i] = h;
    w2l[i] = (__bf16)(v - (float)h);
  }
}

// ---------------------------------------------------------------------------
// Kernel 1: cur1 = x @ W1^T + b1  (M=4096, N=2048, K=1024), bf16-split WMMA.
// Per WG: 16 rows of x, 128 cols (8 waves x one 16-col tile each).
// ---------------------------------------------------------------------------
__global__ __launch_bounds__(256) void gemm1_kernel(
    const float* __restrict__ x, const float* __restrict__ W1,
    const float* __restrict__ b1, float* __restrict__ cur1) {
  constexpr int XS = NI + 8;  // padded bf16 row stride (bank spread)
  __shared__ __bf16 s_xh[16 * XS];
  __shared__ __bf16 s_xl[16 * XS];

  const int tid = threadIdx.x;
  const int mBase = blockIdx.y * 16;
  const int nBlk = blockIdx.x * 128;

  for (int idx = tid; idx < 16 * NI; idx += 256) {
    int row = idx >> 10, col = idx & (NI - 1);
    float xv = x[(size_t)(mBase + row) * NI + col];
    __bf16 h = (__bf16)xv;
    s_xh[row * XS + col] = h;
    s_xl[row * XS + col] = (__bf16)(xv - (float)h);
  }
  __syncthreads();

  const int lane = tid & 31, w = tid >> 5;
  const int n = lane & 15, half = lane >> 4, m = lane & 15;
  const int nT = nBlk + w * 16;
  const float* w1row = W1 + (size_t)(nT + n) * NI;

  v8f c = {};
  for (int kb = 0; kb < NI; kb += 32) {
    v16bf ah, al, bh, bl;
#pragma unroll
    for (int v = 0; v < 8; ++v) {
      // 16-bit A layout: VGPR v holds K pair; lanes 16-31 offset K by +8.
      int kk = kb + ((v >> 2) << 3) + (v << 1) + (half << 3);
      ah[2 * v] = s_xh[m * XS + kk];
      ah[2 * v + 1] = s_xh[m * XS + kk + 1];
      al[2 * v] = s_xl[m * XS + kk];
      al[2 * v + 1] = s_xl[m * XS + kk + 1];
      // 16-bit B layout (32x16): VGPR v holds K pair; lanes 16-31 offset K by +8.
      int kB = kb + ((v >> 2) << 3) + (v << 1) + (half << 3);
      float f0 = w1row[kB], f1 = w1row[kB + 1];
      __bf16 h0 = (__bf16)f0, h1 = (__bf16)f1;
      bh[2 * v] = h0;
      bh[2 * v + 1] = h1;
      bl[2 * v] = (__bf16)(f0 - (float)h0);
      bl[2 * v + 1] = (__bf16)(f1 - (float)h1);
    }
    c = __builtin_amdgcn_wmma_f32_16x16x32_bf16(false, ah, false, bh, (short)0, c, false, false);
    c = __builtin_amdgcn_wmma_f32_16x16x32_bf16(false, ah, false, bl, (short)0, c, false, false);
    c = __builtin_amdgcn_wmma_f32_16x16x32_bf16(false, al, false, bh, (short)0, c, false, false);
  }

  const float bias = b1[nT + n];
#pragma unroll
  for (int r = 0; r < 8; ++r) {
    int mo = mBase + r + (half << 3);  // D layout: VGPR r -> row r (+8 upper half)
    cur1[(size_t)mo * NH + nT + n] = c[r] + bias;
  }
}

// ---------------------------------------------------------------------------
// Kernel 2: fused 25-step LIF loop. One WG owns 16 batch rows for all steps.
// cur1 tile (unpadded, TDM-friendly) + mem1 tile (padded) live in LDS;
// spk1 is formed on the fly from mem1 while building the WMMA A operand;
// mem2 lives in WMMA accumulators; layer-2 uses bf16 hi/lo split of W2.
// ---------------------------------------------------------------------------
__global__ __launch_bounds__(256) void lif_kernel(
    const float* __restrict__ cur1,
    const __bf16* __restrict__ w2h, const __bf16* __restrict__ w2l,
    const float* __restrict__ b2,
    float* __restrict__ outSpk, float* __restrict__ outMem) {
  constexpr int MS = NH + 4;  // padded f32 row stride for mem1 (bank spread)
  __shared__ alignas(16) float s_cur1[16 * NH];  // 128KB, contiguous (DMA dest)
  __shared__ float s_mem1[16 * MS];              // 128.25KB

  const int tid = threadIdx.x;
  const int mBase = blockIdx.x * 16;

  // Zero mem1 ("pre_{-1}": yields pre_0 = cur1).
  for (int idx = tid; idx < 16 * NH; idx += 256) {
    int row = idx >> 11, col = idx & (NH - 1);
    s_mem1[row * MS + col] = 0.0f;
  }

  // Stage cur1 tile (16 x 2048 f32 = 128KB) into LDS.
#if defined(USE_TDM)
  if (tid < 32) {  // wave 0 drives the Tensor Data Mover
    unsigned lds_base = (unsigned)(unsigned long long)(void*)s_cur1;
    unsigned long long ga =
        (unsigned long long)(const void*)(cur1 + (size_t)mBase * NH);
    u32x4 g0;
    i32x8 g1;
    i32x4 gz = {0, 0, 0, 0};
    // D# group 0: count=1 | lds_addr | global_addr[56:0] | type=2
    g0[0] = 1u;
    g0[1] = lds_base;
    g0[2] = (unsigned)ga;
    g0[3] = (unsigned)((ga >> 32) & 0x1FFFFFFull) | (2u << 30);
    // D# group 1: 1-D tile of 32768 4-byte elements.
    g1[0] = (int)(2u << 16);          // data_size=4B, no mask/pad/iterate
    g1[1] = (int)(0x8000u << 16);     // tensor_dim0[15:0]=32768 (bits 63:48)
    g1[2] = (int)(1u << 16);          // tensor_dim0[31:16]=0 | tensor_dim1=1
    g1[3] = (int)(0x8000u << 16);     // tensor_dim1 hi=0 | tile_dim0=32768
    g1[4] = 1;                        // tile_dim1=1, tile_dim2=0
    g1[5] = 32768;                    // tensor_dim0_stride[31:0]
    g1[6] = 0;
    g1[7] = 0;
#if __clang_major__ >= 23
    i32x8 gz8 = {0, 0, 0, 0, 0, 0, 0, 0};
    __builtin_amdgcn_tensor_load_to_lds(g0, g1, gz, gz, gz8, 0);
#else
    __builtin_amdgcn_tensor_load_to_lds(g0, g1, gz, gz, 0);
#endif
    __builtin_amdgcn_s_wait_tensorcnt(0);
  }
#elif defined(USE_ASYNC)
  {
    const float* gsrc = cur1 + (size_t)mBase * NH;
    for (int q = tid * 4; q < 16 * NH; q += 1024) {
      __builtin_amdgcn_global_load_async_to_lds_b128(
          (__attribute__((address_space(1))) void*)(unsigned long long)(const void*)(gsrc + q),
          (__attribute__((address_space(3))) void*)(unsigned)(unsigned long long)(void*)(s_cur1 + q),
          0, 0);
    }
    __builtin_amdgcn_s_wait_asynccnt(0);
  }
#else
  for (int idx = tid; idx < 16 * NH; idx += 256)
    s_cur1[idx] = cur1[(size_t)mBase * NH + idx];
#endif
  __syncthreads();

  const int lane = tid & 31, w = tid >> 5;
  const int n = lane & 15, half = lane >> 4, m = lane & 15;
  const int nt0 = w * 32;  // this wave owns output cols [nt0, nt0+32)
  const float bias0 = b2[nt0 + n];
  const float bias1 = b2[nt0 + 16 + n];
  const __bf16* w2h0 = w2h + (size_t)(nt0 + n) * NH;
  const __bf16* w2l0 = w2l + (size_t)(nt0 + n) * NH;
  const __bf16* w2h1 = w2h + (size_t)(nt0 + 16 + n) * NH;
  const __bf16* w2l1 = w2l + (size_t)(nt0 + 16 + n) * NH;

  v8f acc0 = {}, acc1 = {};  // mem2 (post-reset), persistent across steps

#pragma unroll 1
  for (int t = 0; t < T_STEPS; ++t) {
    // Layer-1 LIF advance: pre_t = beta*(pre_{t-1} - spk_{t-1}) + cur1
    for (int idx = tid; idx < 16 * NH; idx += 256) {
      int row = idx >> 11, col = idx & (NH - 1);
      int ad = row * MS + col;
      float pre = s_mem1[ad];
      float spk = pre > THRESH ? 1.0f : 0.0f;
      s_mem1[ad] = BETA * (pre - spk) + s_cur1[idx];
    }
    __syncthreads();

    // Layer-2: cur2 = spk1 @ W2^T + b2 ; seed C with beta*mem2 + b2.
    v8f c0, c1;
#pragma unroll
    for (int r = 0; r < 8; ++r) {
      c0[r] = BETA * acc0[r] + bias0;
      c1[r] = BETA * acc1[r] + bias1;
    }

    for (int kb = 0; kb < NH; kb += 32) {
      __builtin_prefetch(w2h0 + kb + 256);  // global_prefetch_b8
      __builtin_prefetch(w2h1 + kb + 256);
      v16bf av, bh0, bl0, bh1, bl1;
#pragma unroll
      for (int v = 0; v < 8; ++v) {
        int kk = kb + ((v >> 2) << 3) + (v << 1) + (half << 3);
        float p0 = s_mem1[m * MS + kk];
        float p1 = s_mem1[m * MS + kk + 1];
        av[2 * v]     = p0 > THRESH ? (__bf16)1.0f : (__bf16)0.0f;  // exact
        av[2 * v + 1] = p1 > THRESH ? (__bf16)1.0f : (__bf16)0.0f;
        int kB = kb + ((v >> 2) << 3) + (v << 1) + (half << 3);
        bh0[2 * v] = w2h0[kB]; bh0[2 * v + 1] = w2h0[kB + 1];
        bl0[2 * v] = w2l0[kB]; bl0[2 * v + 1] = w2l0[kB + 1];
        bh1[2 * v] = w2h1[kB]; bh1[2 * v + 1] = w2h1[kB + 1];
        bl1[2 * v] = w2l1[kB]; bl1[2 * v + 1] = w2l1[kB + 1];
      }
      c0 = __builtin_amdgcn_wmma_f32_16x16x32_bf16(false, av, false, bh0, (short)0, c0, false, false);
      c0 = __builtin_amdgcn_wmma_f32_16x16x32_bf16(false, av, false, bl0, (short)0, c0, false, false);
      c1 = __builtin_amdgcn_wmma_f32_16x16x32_bf16(false, av, false, bh1, (short)0, c1, false, false);
      c1 = __builtin_amdgcn_wmma_f32_16x16x32_bf16(false, av, false, bl1, (short)0, c1, false, false);
    }

    // Layer-2 spike/reset + record, keep post-reset mem2 in accumulators.
    const size_t tOff = (size_t)t * Bsz * NO;
#pragma unroll
    for (int r = 0; r < 8; ++r) {
      int mo = mBase + r + (half << 3);
      size_t o0 = tOff + (size_t)mo * NO + nt0 + n;
      size_t o1 = o0 + 16;
      float p0 = c0[r], s0 = p0 > THRESH ? 1.0f : 0.0f, q0 = p0 - s0;
      float p1 = c1[r], s1 = p1 > THRESH ? 1.0f : 0.0f, q1 = p1 - s1;
      outSpk[o0] = s0; outMem[o0] = q0;
      outSpk[o1] = s1; outMem[o1] = q1;
      acc0[r] = q0; acc1[r] = q1;
    }
    __syncthreads();  // protect s_mem1 reads vs next step's writes
  }
}

// ---------------------------------------------------------------------------
extern "C" void kernel_launch(void* const* d_in, const int* in_sizes, int n_in,
                              void* d_out, int out_size, void* d_ws, size_t ws_size,
                              hipStream_t stream) {
  const float* x  = (const float*)d_in[0];
  const float* W1 = (const float*)d_in[1];
  const float* b1 = (const float*)d_in[2];
  const float* W2 = (const float*)d_in[3];
  const float* b2 = (const float*)d_in[4];

  float* out = (float*)d_out;
  float* outSpk = out;                               // [25,4096,256]
  float* outMem = out + (size_t)T_STEPS * Bsz * NO;  // [25,4096,256]

  // Workspace: cur1 f32 (33.5 MB) | W2 hi bf16 (1 MB) | W2 lo bf16 (1 MB)
  char* ws = (char*)d_ws;
  float* cur1 = (float*)ws;
  __bf16* w2h = (__bf16*)(ws + (size_t)Bsz * NH * sizeof(float));
  __bf16* w2l = w2h + (size_t)NO * NH;

  split_w2_kernel<<<(NO * NH + 255) / 256, 256, 0, stream>>>(W2, w2h, w2l);
  gemm1_kernel<<<dim3(NH / 128, Bsz / 16), 256, 0, stream>>>(x, W1, b1, cur1);
  lif_kernel<<<Bsz / 16, 256, 0, stream>>>(cur1, w2h, w2l, b2, outSpk, outMem);
}
